// DotProductAttention_28759101014802
// MI455X (gfx1250) — matrix operand order repaired and checked by reading, not compiled, not run on hardware
//
#include <hip/hip_runtime.h>

// ---------------- problem constants ----------------
#define MD   1024   // MODEL_DIM == DK == DV
#define BB   4
#define SS   2048
#define MTOT (BB*SS) // 8192 rows across batches
#define LDP  40     // padded LDS row (bf16 elems) for a 32-wide k-slice

typedef __attribute__((ext_vector_type(16))) __bf16 v16bf;
typedef __attribute__((ext_vector_type(8)))  __bf16 v8bf;
typedef __attribute__((ext_vector_type(8)))  float  v8f;
typedef __attribute__((ext_vector_type(4)))  float  v4f;
typedef __attribute__((ext_vector_type(4)))  int    v4i;

#define WMMA_BF16(a, b, c) \
    __builtin_amdgcn_wmma_f32_16x16x32_bf16(false, (a), false, (b), (short)0, (c), false, false)

// ---------------- WMMA fragment loaders (wave32, 16x16x32 bf16) ----------------
// A 16x32 bf16: lane m=lane&15, kb=(lane>>4)*8; j0..7 -> K=kb..kb+7, j8..15 -> K=kb+16..kb+23
__device__ __forceinline__ v16bf frag_a_bf16(const __bf16* base, int ld) {
    int lane = threadIdx.x & 31;
    const __bf16* p = base + (lane & 15) * ld + ((lane >> 4) * 8);
    v8bf lo = *(const v8bf*)(p);
    v8bf hi = *(const v8bf*)(p + 16);
    return __builtin_shufflevector(lo, hi, 0,1,2,3,4,5,6,7,8,9,10,11,12,13,14,15);
}

// A 16x32 from fp32 memory, converted inline to bf16
__device__ __forceinline__ v16bf frag_a_f32(const float* __restrict__ base, int ld) {
    int lane = threadIdx.x & 31;
    const float* p = base + (lane & 15) * ld + ((lane >> 4) * 8);
    v4f x0 = *(const v4f*)(p);
    v4f x1 = *(const v4f*)(p + 4);
    v4f y0 = *(const v4f*)(p + 16);
    v4f y1 = *(const v4f*)(p + 20);
    v16bf a;
    a[0]=(__bf16)x0[0]; a[1]=(__bf16)x0[1]; a[2]=(__bf16)x0[2]; a[3]=(__bf16)x0[3];
    a[4]=(__bf16)x1[0]; a[5]=(__bf16)x1[1]; a[6]=(__bf16)x1[2]; a[7]=(__bf16)x1[3];
    a[8]=(__bf16)y0[0]; a[9]=(__bf16)y0[1]; a[10]=(__bf16)y0[2]; a[11]=(__bf16)y0[3];
    a[12]=(__bf16)y1[0]; a[13]=(__bf16)y1[1]; a[14]=(__bf16)y1[2]; a[15]=(__bf16)y1[3];
    return a;
}

// B 32x16 bf16 from TRANSPOSED storage bt[n][k]: lane n=lane&15, kb=(lane>>4)*8; j0..7 -> K=kb..kb+7, j8..15 -> K=kb+16..kb+23
__device__ __forceinline__ v16bf frag_bt_bf16(const __bf16* bt, int ld) {
    int lane = threadIdx.x & 31;
    const __bf16* p = bt + (lane & 15) * ld + ((lane >> 4) * 8);
    v8bf lo = *(const v8bf*)(p);
    v8bf hi = *(const v8bf*)(p + 16);
    return __builtin_shufflevector(lo, hi, 0,1,2,3,4,5,6,7,8,9,10,11,12,13,14,15);
}

// ---------------- async global->LDS copy (16B per lane) ----------------
__device__ __forceinline__ void cp_async16(const __bf16* g, __bf16* l) {
#if __has_builtin(__builtin_amdgcn_global_load_async_to_lds_b128)
    __builtin_amdgcn_global_load_async_to_lds_b128(
        (__attribute__((address_space(1))) v4i*)(g),
        (__attribute__((address_space(3))) v4i*)(l), 0, 0);
#else
    *(v8bf*)l = *(const v8bf*)g;   // synchronous fallback
#endif
}

__device__ __forceinline__ void wait_async_zero() {
#if __has_builtin(__builtin_amdgcn_s_wait_asynccnt)
    __builtin_amdgcn_s_wait_asynccnt(0);
#else
    asm volatile("s_wait_asynccnt 0" ::: "memory");
#endif
}

// ---------------- kernel 1: weight convert + transpose (fp32 [d][n] -> bf16 [n][d]) ----------------
__global__ __launch_bounds__(256)
void wconv_kernel(const float* __restrict__ W, __bf16* __restrict__ Wt) {
    __shared__ float tile[32][33];
    int tx = threadIdx.x, ty = threadIdx.y;            // (32, 8)
    int n = blockIdx.x * 32 + tx;
    #pragma unroll
    for (int i = 0; i < 32; i += 8)
        tile[ty + i][tx] = W[(size_t)(blockIdx.y * 32 + ty + i) * MD + n];
    __syncthreads();
    #pragma unroll
    for (int i = 0; i < 32; i += 8)
        Wt[(size_t)(blockIdx.x * 32 + ty + i) * MD + (blockIdx.y * 32 + tx)] =
            (__bf16)tile[tx][ty + i];
}

// ---------------- kernel 2: projection GEMM  Out = X @ W + bias (bf16 out) ----------------
// Register-blocked: 8 waves as 2(M)x4(N); per-wave 32x64; WG tile 64x256.
__global__ __launch_bounds__(256)
void proj_kernel(const float* __restrict__ X, const __bf16* __restrict__ Wt,
                 const float* __restrict__ bias, __bf16* __restrict__ Out,
                 int transpose_out) {
    int w = threadIdx.x >> 5, lane = threadIdx.x & 31;
    int mi = w >> 2, ni = w & 3;
    int mW = blockIdx.x * 64 + mi * 32;
    int nW = blockIdx.y * 256 + ni * 64;
    const float* a0 = X + (size_t)mW * MD;
    const float* a1 = a0 + (size_t)16 * MD;
    v8f acc[2][4] = {};
    for (int k0 = 0; k0 < MD; k0 += 32) {
        __builtin_prefetch(a0 + k0 + 256, 0, 1);
        __builtin_prefetch(a1 + k0 + 256, 0, 1);
        v16bf A0 = frag_a_f32(a0 + k0, MD);
        v16bf A1 = frag_a_f32(a1 + k0, MD);
        v16bf Bf[4];
        #pragma unroll
        for (int j = 0; j < 4; ++j)
            Bf[j] = frag_bt_bf16(Wt + (size_t)(nW + j * 16) * MD + k0, MD);
        #pragma unroll
        for (int j = 0; j < 4; ++j) {
            acc[0][j] = WMMA_BF16(A0, Bf[j], acc[0][j]);
            acc[1][j] = WMMA_BF16(A1, Bf[j], acc[1][j]);
        }
    }
    int nl = lane & 15, mb = (lane >> 4) * 8;
    #pragma unroll
    for (int i = 0; i < 2; ++i) {
        #pragma unroll
        for (int j = 0; j < 4; ++j) {
            int n = nW + j * 16 + nl;
            int m = mW + i * 16 + mb;
            float bv = bias[n];
            if (!transpose_out) {
                #pragma unroll
                for (int r = 0; r < 8; ++r)
                    Out[(size_t)(m + r) * MD + n] = (__bf16)(acc[i][j][r] + bv);
            } else {
                int bat = m >> 11;           // / SS
                int s   = m & (SS - 1);
                v8bf o;
                #pragma unroll
                for (int r = 0; r < 8; ++r) o[r] = (__bf16)(acc[i][j][r] + bv);
                *(v8bf*)(Out + ((size_t)bat * MD + n) * SS + s) = o;
            }
        }
    }
}

// ---------------- kernel 3: generic NT-GEMM with async-LDS double buffering ----------------
// C[M x N] f32 = A[M x kdim] (bf16, lda) @ B^T where Bt[N x kdim] (bf16, ldb).
// Used for scores (A=Q, Bt=K) and for P@V (A=P, Bt=Vt). WG tile 64M x 256N.
__global__ __launch_bounds__(256)
void gemm_nt_kernel(const __bf16* __restrict__ A, const __bf16* __restrict__ Bt,
                    float* __restrict__ C, int lda, int ldb, int ldc, int kdim) {
    __shared__ __bf16 As[2][64][LDP];
    __shared__ __bf16 Bs[2][256][LDP];
    int t = threadIdx.x;
    int w = t >> 5, lane = t & 31;
    int mi = w >> 2, ni = w & 3;
    int m0 = blockIdx.x * 64;
    int n0 = blockIdx.y * 256;
    const __bf16* Ag = A  + (size_t)m0 * lda;
    const __bf16* Bg = Bt + (size_t)n0 * ldb;
    v8f acc[2][4] = {};

    auto stage = [&](int buf, int d0) {
        // A tile: 64 rows x 32k -> 256 chunks of 16B, one per thread
        { int row = t >> 2, seg = t & 3;
          cp_async16(Ag + (size_t)row * lda + d0 + seg * 8, &As[buf][row][seg * 8]); }
        // B tile: 256 rows x 32k -> 1024 chunks, four per thread
        #pragma unroll
        for (int i = 0; i < 4; ++i) {
            int c = t + 256 * i, row = c >> 2, seg = c & 3;
            cp_async16(Bg + (size_t)row * ldb + d0 + seg * 8, &Bs[buf][row][seg * 8]);
        }
    };

    stage(0, 0);
    int niter = kdim >> 5;
    for (int it = 0; it < niter; ++it) {
        int cur = it & 1;
        wait_async_zero();
        __syncthreads();
        if (it + 1 < niter) stage(cur ^ 1, (it + 1) * 32);
        v16bf A0 = frag_a_bf16(&As[cur][mi * 32][0],      LDP);
        v16bf A1 = frag_a_bf16(&As[cur][mi * 32 + 16][0], LDP);
        v16bf Bf[4];
        #pragma unroll
        for (int j = 0; j < 4; ++j)
            Bf[j] = frag_bt_bf16(&Bs[cur][ni * 64 + j * 16][0], LDP);
        #pragma unroll
        for (int j = 0; j < 4; ++j) {
            acc[0][j] = WMMA_BF16(A0, Bf[j], acc[0][j]);
            acc[1][j] = WMMA_BF16(A1, Bf[j], acc[1][j]);
        }
    }
    int nl = lane & 15, mb = (lane >> 4) * 8;
    #pragma unroll
    for (int i = 0; i < 2; ++i)
        #pragma unroll
        for (int j = 0; j < 4; ++j) {
            int n = n0 + ni * 64 + j * 16 + nl;
            int m = m0 + mi * 32 + i * 16 + mb;
            #pragma unroll
            for (int r = 0; r < 8; ++r)
                C[(size_t)(m + r) * ldc + n] = acc[i][j][r];
        }
}

// ---------------- kernel 4: row softmax (scale 1/32), write P bf16 ----------------
__global__ __launch_bounds__(256)
void softmax_kernel(const float* __restrict__ Sc, __bf16* __restrict__ P) {
    const float scale = 0.03125f;              // 1/sqrt(1024)
    int row  = blockIdx.x * 8 + (threadIdx.x >> 5);
    int lane = threadIdx.x & 31;
    const float* r = Sc + (size_t)row * SS;
    float vals[SS / 32];
    float m = -3.4e38f;
    #pragma unroll 8
    for (int i = 0; i < SS / 32; ++i) {
        float v = r[lane + i * 32] * scale;
        vals[i] = v;
        m = fmaxf(m, v);
    }
    #pragma unroll
    for (int off = 16; off > 0; off >>= 1) m = fmaxf(m, __shfl_xor(m, off, 32));
    float sum = 0.f;
    #pragma unroll 8
    for (int i = 0; i < SS / 32; ++i) {
        float e = __expf(vals[i] - m);
        vals[i] = e;
        sum += e;
    }
    #pragma unroll
    for (int off = 16; off > 0; off >>= 1) sum += __shfl_xor(sum, off, 32);
    float inv = 1.0f / sum;
    __bf16* p = P + (size_t)row * SS;
    #pragma unroll 8
    for (int i = 0; i < SS / 32; ++i) p[lane + i * 32] = (__bf16)(vals[i] * inv);
}

// ---------------- host-side launcher ----------------
extern "C" void kernel_launch(void* const* d_in, const int* in_sizes, int n_in,
                              void* d_out, int out_size, void* d_ws, size_t ws_size,
                              hipStream_t stream) {
    (void)in_sizes; (void)n_in; (void)out_size; (void)ws_size;
    const float* q  = (const float*)d_in[0];
    const float* k  = (const float*)d_in[1];
    const float* v  = (const float*)d_in[2];
    const float* Wq = (const float*)d_in[3];
    const float* Wk = (const float*)d_in[4];
    const float* Wv = (const float*)d_in[5];
    const float* bq = (const float*)d_in[6];
    const float* bk = (const float*)d_in[7];
    const float* bv = (const float*)d_in[8];
    float* out = (float*)d_out;

    // workspace carve-out (bytes)
    char* ws = (char*)d_ws;
    const size_t WSZ = (size_t)MD * MD * 2;           // 2 MB per bf16 weight
    const size_t QSZ = (size_t)MTOT * MD * 2;         // 16 MB
    __bf16* Wtq = (__bf16*)(ws);
    __bf16* Wtk = (__bf16*)(ws + WSZ);
    __bf16* Wtv = (__bf16*)(ws + 2 * WSZ);
    __bf16* Qb  = (__bf16*)(ws + 3 * WSZ);
    __bf16* Kb  = (__bf16*)(ws + 3 * WSZ + QSZ);
    __bf16* Vt  = (__bf16*)(ws + 3 * WSZ + 2 * QSZ);          // [B][MD][SS]
    float*  Sc  = (float*) (ws + 3 * WSZ + 3 * QSZ);          // 16 MB, reused per batch
    __bf16* P   = (__bf16*)(ws + 3 * WSZ + 3 * QSZ + (size_t)SS * SS * 4); // 8 MB

    // 1) weights -> bf16 transposed
    hipLaunchKernelGGL(wconv_kernel, dim3(MD / 32, MD / 32), dim3(32, 8), 0, stream, Wq, Wtq);
    hipLaunchKernelGGL(wconv_kernel, dim3(MD / 32, MD / 32), dim3(32, 8), 0, stream, Wk, Wtk);
    hipLaunchKernelGGL(wconv_kernel, dim3(MD / 32, MD / 32), dim3(32, 8), 0, stream, Wv, Wtv);

    // 2) projections (register-blocked WMMA GEMM)
    dim3 pg(MTOT / 64, MD / 256);
    hipLaunchKernelGGL(proj_kernel, pg, dim3(256), 0, stream, q, Wtq, bq, Qb, 0);
    hipLaunchKernelGGL(proj_kernel, pg, dim3(256), 0, stream, k, Wtk, bk, Kb, 0);
    hipLaunchKernelGGL(proj_kernel, pg, dim3(256), 0, stream, v, Wtv, bv, Vt, 1);

    // 3) per-batch attention (sequential launches reuse Sc/P scratch safely)
    for (int b = 0; b < BB; ++b) {
        hipLaunchKernelGGL(gemm_nt_kernel, dim3(SS / 64, SS / 256), dim3(256), 0, stream,
                           Qb + (size_t)b * SS * MD, Kb + (size_t)b * SS * MD, Sc,
                           MD, MD, SS, MD);
        hipLaunchKernelGGL(softmax_kernel, dim3(SS / 8), dim3(256), 0, stream, Sc, P);
        hipLaunchKernelGGL(gemm_nt_kernel, dim3(SS / 64, MD / 256), dim3(256), 0, stream,
                           P, Vt + (size_t)b * MD * SS, out + (size_t)b * SS * MD,
                           SS, SS, MD, SS);
    }
}
